// BrainGNN_20564303414149
// MI455X (gfx1250) — matrix operand order repaired and checked by reading, not compiled, not run on hardware
//
#include <hip/hip_runtime.h>
#include <hip/hip_bf16.h>

// ---------------- problem constants ----------------
#define B_    1024
#define NODE  112
#define KP1   128     // NODE padded to K multiple of 32
#define P1    56      // nodes kept by pool1
#define P1P   64      // padded
#define P2    28      // nodes kept by pool2
#define L1    512
#define L2    512
#define L3    256
#define CL    8

// ---------------- WMMA types (CDNA5, wave32) ----------------
typedef __bf16 v16bf __attribute__((ext_vector_type(16)));
typedef float  v8f   __attribute__((ext_vector_type(8)));
typedef unsigned short u16x8 __attribute__((ext_vector_type(8)));

union FragBF { v16bf v; u16x8 h[2]; };

__device__ __forceinline__ unsigned short f2bf(float x) {
  union { float f; unsigned u; } c; c.f = x;
  return (unsigned short)((c.u + 0x7FFFu + ((c.u >> 16) & 1u)) >> 16); // RNE
}
__device__ __forceinline__ float bf2f(unsigned short h) {
  union { unsigned u; float f; } c; c.u = ((unsigned)h) << 16;
  return c.f;
}

// One 16x64 output strip per wave: 4 accumulators share each A fragment
// (4 WMMAs per 1KB A-fragment load -> 4x arithmetic intensity vs 1 tile).
// All B fragments of a k-step are loaded BEFORE the 4 WMMAs so the scheduler
// can retire them with descending s_wait_loadcnt instead of draining to 0
// before every MMA. A: row-major [16 x K] tile (lda elems). Bt: B stored
// TRANSPOSED as [N][K] so a B fragment (same K order as A, ISA 7.12.2 table) is two
// contiguous 16B runs per lane (K+0..7, K+16..23).
__device__ __forceinline__ void wmma_strip4(const unsigned short* __restrict__ A, int lda,
                                            const unsigned short* __restrict__ Bt, int ldb,
                                            int K, int lane, v8f acc[4]) {
  const int r    = lane & 15;
  const int aklo = (lane < 16) ? 0 : 8;   // A: lanes16-31 start at K=8 (ISA table)
  const int bklo = (lane < 16) ? 0 : 8;   // B: lanes16-31 start at K=8 (same as A)
  const unsigned short* ap = A  + (size_t)r * lda + aklo;
  const unsigned short* b0 = Bt + (size_t)r * ldb + bklo;
  for (int k0 = 0; k0 < K; k0 += 32) {
    FragBF a, b[4];
    a.h[0] = *(const u16x8*)(ap + k0);        // A elems 0..7  -> K=aklo+0..7
    a.h[1] = *(const u16x8*)(ap + k0 + 16);   // A elems 8..15 -> K=aklo+16..23
#pragma unroll
    for (int j = 0; j < 4; ++j) {             // issue all B loads up front
      const unsigned short* bp = b0 + (size_t)(j * 16) * ldb + k0;
      b[j].h[0] = *(const u16x8*)(bp);        // B elems 0..7  -> K=bklo+0..7
      b[j].h[1] = *(const u16x8*)(bp + 16);   // B elems 8..15 -> K=bklo+16..23
    }
    if (k0 + 32 < K) {                        // uniform branch, EXEC untouched
      __builtin_prefetch(ap + k0 + 64, 0, 1); // global_prefetch_b8
#pragma unroll
      for (int j = 0; j < 4; ++j)             // each column = distinct lines
        __builtin_prefetch(b0 + (size_t)(j * 16) * ldb + k0 + 64, 0, 1);
    }
#pragma unroll
    for (int j = 0; j < 4; ++j)
      acc[j] = __builtin_amdgcn_wmma_f32_16x16x32_bf16(false, a.v, false, b[j].v,
                                                       (short)0, acc[j], false, false);
  }
}

// ---------------- stage-1 weight diag: W1d[n,o] = (a1@b1)[n, n*L1+o] + bias1[o]
__global__ void k_w1diag(const float* __restrict__ alpha1, const float* __restrict__ beta1,
                         const float* __restrict__ bias1,
                         float* __restrict__ W1df, unsigned short* __restrict__ Bt1) {
  int t = blockIdx.x * blockDim.x + threadIdx.x;   // L1*KP1 threads
  int o = t >> 7, n = t & (KP1 - 1);
  float v = 0.f;
  if (n < NODE) {
    float s = bias1[o];
    for (int c = 0; c < CL; ++c)
      s += alpha1[n * CL + c] * beta1[(size_t)c * (NODE * L1) + n * L1 + o];
    W1df[n * L1 + o] = s;
    v = s;
  }
  Bt1[o * KP1 + n] = f2bf(v);                      // transposed + K padded to 128
}

// ---------------- x -> bf16, K padded 112->128 ----------------
__global__ void k_cvt_x(const float* __restrict__ x, unsigned short* __restrict__ Xbf) {
  int t = blockIdx.x * blockDim.x + threadIdx.x;   // B_*NODE*KP1
  int k = t & (KP1 - 1);
  int row = t >> 7;
  Xbf[t] = (k < NODE) ? f2bf(x[(size_t)row * NODE + k]) : (unsigned short)0;
}

// ---------------- stage-1 GEMM: h1 = relu(x@W1d + W1d) ----------------
// Block = 8 waves, each wave a 16x64 strip -> block covers 128(M) x 64(N).
__global__ void k_gemm_s1(const unsigned short* __restrict__ Xbf,
                          const unsigned short* __restrict__ Bt1,
                          const float* __restrict__ W1df, unsigned short* __restrict__ h1) {
  int wave = threadIdx.x >> 5, lane = threadIdx.x & 31;
  int tileM = blockIdx.y * 128 + wave * 16;
  int tileN = blockIdx.x * 64;
  v8f acc[4] = {};
  wmma_strip4(Xbf + (size_t)tileM * KP1, KP1,
              Bt1 + (size_t)tileN * KP1, KP1, KP1, lane, acc);
  // cheap row % NODE: one mod per thread, then conditional subtract per row
  int m0 = tileM % NODE;                       // tile-uniform
  int roff = (lane < 16) ? 0 : 8;
#pragma unroll
  for (int j = 0; j < 4; ++j) {
    int col = tileN + j * 16 + (lane & 15);
#pragma unroll
    for (int r = 0; r < 8; ++r) {
      int row = tileM + roff + r;              // row = b*NODE + m
      int m = m0 + roff + r;
      if (m >= NODE) m -= NODE;                // offset <= 23 < NODE, one sub
      float v = acc[j][r] + W1df[m * L1 + col];
      h1[(size_t)row * L1 + col] = f2bf(fmaxf(v, 0.f));
    }
  }
}

// ---------------- pool: sigmoid scores + rank-select (== top_k order) ------
__global__ void k_pool1(const unsigned short* __restrict__ h1,
                        const float* __restrict__ pw, const float* __restrict__ pb,
                        int* __restrict__ idx1, float* __restrict__ val1) {
  __shared__ float sc[NODE];
  int b = blockIdx.x, t = threadIdx.x;
  if (t < NODE) {
    const unsigned short* hr = h1 + ((size_t)b * NODE + t) * L1;
    float s = pb[0];
    for (int k = 0; k < L1; ++k) s += bf2f(hr[k]) * pw[k];
    sc[t] = 1.f / (1.f + __expf(-s));
  }
  __syncthreads();
  if (t < NODE) {
    float my = sc[t]; int rank = 0;
    for (int m = 0; m < NODE; ++m) {
      float o = sc[m];
      rank += (o > my) || (o == my && m < t);          // descending, stable
    }
    if (rank < P1) { idx1[b * P1 + rank] = t; val1[b * P1 + rank] = my; }
  }
}

// ---------------- gather1: hg = h1[idx]*val ; adjg = renorm(x[idx,idx]) pad64
__global__ void k_gather1(const unsigned short* __restrict__ h1, const float* __restrict__ x,
                          const int* __restrict__ idx1, const float* __restrict__ val1,
                          unsigned short* __restrict__ hg, unsigned short* __restrict__ adjg) {
  __shared__ int sidx[P1];
  __shared__ float sval[P1];
  int b = blockIdx.x, t = threadIdx.x;
  if (t < P1) { sidx[t] = idx1[b * P1 + t]; sval[t] = val1[b * P1 + t]; }
  __syncthreads();
  for (int e = t; e < P1 * L1; e += blockDim.x) {
    int j = e >> 9, o = e & (L1 - 1);
    hg[((size_t)b * P1 + j) * L1 + o] =
        f2bf(bf2f(h1[((size_t)b * NODE + sidx[j]) * L1 + o]) * sval[j]);
  }
  if (t < P1P) {
    unsigned short* arow = adjg + ((size_t)b * P1P + t) * P1P;
    if (t < P1) {
      const float* xr = x + ((size_t)b * NODE + sidx[t]) * NODE;
      float s = 0.f;
      for (int k = 0; k < P1; ++k) s += xr[sidx[k]];
      float inv = 1.f / s;
      for (int k = 0; k < P1; ++k) arow[k] = f2bf(xr[sidx[k]] * inv);
      for (int k = P1; k < P1P; ++k) arow[k] = 0;
    } else {
      for (int k = 0; k < P1P; ++k) arow[k] = 0;
    }
  }
}

// ---------------- per-node weights, transposed: w2t[n][o][i] ----------------
__global__ void k_w2t(const float* __restrict__ alpha2, const float* __restrict__ beta2,
                      unsigned short* __restrict__ w2t) {
  int t = blockIdx.x * blockDim.x + threadIdx.x;   // P1*L2*L1
  int i = t & (L1 - 1);
  int o = (t >> 9) & (L2 - 1);
  int n = t >> 18;
  float s = 0.f;
  for (int c = 0; c < CL; ++c)
    s += alpha2[n * CL + c] * beta2[(size_t)c * (L1 * L2) + i * L2 + o];
  w2t[((size_t)n * L2 + o) * L1 + i] = f2bf(s);
}

// ---------------- stage-2 per-node GEMM: h2[:,n,:] = hg[:,n,:] @ w2[n] + bias2
__global__ void k_gemm_s2a(const unsigned short* __restrict__ hg,
                           const unsigned short* __restrict__ w2t,
                           const float* __restrict__ bias2,
                           unsigned short* __restrict__ h2, unsigned short* __restrict__ h2t) {
  int wave = threadIdx.x >> 5, lane = threadIdx.x & 31;
  int n = blockIdx.z;
  int tileM = blockIdx.y * 128 + wave * 16;            // batch dim
  int tileN = blockIdx.x * 64;
  const unsigned short* A  = hg  + ((size_t)tileM * P1 + n) * L1;  // row stride P1*L1
  const unsigned short* Bt = w2t + ((size_t)n * L2 + tileN) * L1;
  v8f acc[4] = {};
  wmma_strip4(A, P1 * L1, Bt, L1, L1, lane, acc);
  int roff = (lane < 16) ? 0 : 8;
#pragma unroll
  for (int j = 0; j < 4; ++j) {
    int col = tileN + j * 16 + (lane & 15);
    float bz = bias2[col];
#pragma unroll
    for (int r = 0; r < 8; ++r) {
      int row = tileM + roff + r;                      // batch index
      unsigned short hv = f2bf(acc[j][r] + bz);
      h2 [((size_t)row * P1 + n) * L2 + col] = hv;     // for the "+h2" addend
      h2t[((size_t)row * L2 + col) * P1P + n] = hv;    // B operand of s2b
    }
  }
}

// ---------------- stage-2 propagate: hB = relu(adjg @ h2 + h2) per batch ----
// Block = 8 waves as 4(M) x 2(N-strips of 64) -> covers 64(M) x 128(N).
__global__ void k_gemm_s2b(const unsigned short* __restrict__ adjg,
                           const unsigned short* __restrict__ h2t,
                           const unsigned short* __restrict__ h2,
                           unsigned short* __restrict__ hB) {
  int wave = threadIdx.x >> 5, lane = threadIdx.x & 31;
  int b = blockIdx.y;
  int tileM = (wave >> 1) * 16;                        // 0..48 (64 padded rows)
  int tileN = blockIdx.x * 128 + (wave & 1) * 64;
  const unsigned short* A  = adjg + (size_t)b * P1P * P1P + (size_t)tileM * P1P;
  const unsigned short* Bt = h2t  + (size_t)b * L2 * P1P + (size_t)tileN * P1P;
  v8f acc[4] = {};
  wmma_strip4(A, P1P, Bt, P1P, P1P, lane, acc);
  int roff = (lane < 16) ? 0 : 8;
#pragma unroll
  for (int j = 0; j < 4; ++j) {
    int col = tileN + j * 16 + (lane & 15);
#pragma unroll
    for (int r = 0; r < 8; ++r) {
      int row = tileM + roff + r;
      if (row < P1) {
        float v = acc[j][r] + bf2f(h2[((size_t)b * P1 + row) * L2 + col]);
        hB[((size_t)b * P1 + row) * L2 + col] = f2bf(fmaxf(v, 0.f));
      }
    }
  }
}

// ---------------- pool2 + node-mean fused -> pooled[b, 512] ----------------
__global__ void k_pool2(const unsigned short* __restrict__ hB,
                        const float* __restrict__ pw, const float* __restrict__ pb,
                        float* __restrict__ pooled) {
  __shared__ float sc[P1];
  __shared__ int   sidx[P2];
  __shared__ float sval[P2];
  int b = blockIdx.x, t = threadIdx.x;
  if (t < P1) {
    const unsigned short* hr = hB + ((size_t)b * P1 + t) * L2;
    float s = pb[0];
    for (int k = 0; k < L2; ++k) s += bf2f(hr[k]) * pw[k];
    sc[t] = 1.f / (1.f + __expf(-s));
  }
  __syncthreads();
  if (t < P1) {
    float my = sc[t]; int rank = 0;
    for (int m = 0; m < P1; ++m) {
      float o = sc[m];
      rank += (o > my) || (o == my && m < t);
    }
    if (rank < P2) { sidx[rank] = t; sval[rank] = my; }
  }
  __syncthreads();
  for (int o = t; o < L2; o += blockDim.x) {
    float s = 0.f;
    for (int j = 0; j < P2; ++j)
      s += bf2f(hB[((size_t)b * P1 + sidx[j]) * L2 + o]) * sval[j];
    pooled[(size_t)b * L2 + o] = s * (1.f / (float)P2);
  }
}

// ---------------- head: out = relu(pooled@fc1^T + b1) @ fc2^T + b2 ----------
__global__ void k_head(const float* __restrict__ pooled,
                       const float* __restrict__ fc1_w, const float* __restrict__ fc1_b,
                       const float* __restrict__ fc2_w, const float* __restrict__ fc2_b,
                       float* __restrict__ out) {
  __shared__ float hin[L2];
  __shared__ float red[L3];
  int b = blockIdx.x, t = threadIdx.x;     // L3 threads
  for (int o = t; o < L2; o += blockDim.x) hin[o] = pooled[(size_t)b * L2 + o];
  __syncthreads();
  float a = fc1_b[t];
  for (int k = 0; k < L2; ++k) a += hin[k] * fc1_w[t * L2 + k];
  red[t] = fmaxf(a, 0.f) * fc2_w[t];
  __syncthreads();
  for (int s = L3 / 2; s > 0; s >>= 1) {
    if (t < s) red[t] += red[t + s];
    __syncthreads();
  }
  if (t == 0) out[b] = red[0] + fc2_b[0];
}

// ---------------- launch ----------------
extern "C" void kernel_launch(void* const* d_in, const int* in_sizes, int n_in,
                              void* d_out, int out_size, void* d_ws, size_t ws_size,
                              hipStream_t stream) {
  (void)in_sizes; (void)n_in; (void)out_size; (void)ws_size;
  const float* x      = (const float*)d_in[0];
  const float* alpha1 = (const float*)d_in[1];
  const float* beta1  = (const float*)d_in[2];
  const float* bias1  = (const float*)d_in[3];
  const float* p1w    = (const float*)d_in[4];
  const float* p1b    = (const float*)d_in[5];
  const float* alpha2 = (const float*)d_in[6];
  const float* beta2  = (const float*)d_in[7];
  const float* bias2  = (const float*)d_in[8];
  const float* p2w    = (const float*)d_in[9];
  const float* p2b    = (const float*)d_in[10];
  const float* fc1_w  = (const float*)d_in[11];
  const float* fc1_b  = (const float*)d_in[12];
  const float* fc2_w  = (const float*)d_in[13];
  const float* fc2_b  = (const float*)d_in[14];
  float* out = (float*)d_out;

  // workspace carving (256B aligned)
  char* p = (char*)d_ws;
  auto carve = [&](size_t bytes) -> void* {
    void* r = (void*)p;
    p += (bytes + 255) & ~(size_t)255;
    return r;
  };
  float*          W1df = (float*)carve((size_t)NODE * L1 * 4);
  unsigned short* Bt1  = (unsigned short*)carve((size_t)L1 * KP1 * 2);
  // Xbf (s1 A-operand) is dead after s1 GEMM; reuse the region for w2t.
  unsigned short* Xbf  = (unsigned short*)carve((size_t)B_ * NODE * KP1 * 2); // 29.4 MB
  unsigned short* w2t  = Xbf;
  // h1 is dead after gather1; reuse its region for h2 (58.7 MB <= 117.4 MB).
  unsigned short* h1   = (unsigned short*)carve((size_t)B_ * NODE * L1 * 2);  // 117.4 MB
  unsigned short* h2   = h1;
  int*            idx1 = (int*)carve((size_t)B_ * P1 * 4);
  float*          val1 = (float*)carve((size_t)B_ * P1 * 4);
  unsigned short* hg   = (unsigned short*)carve((size_t)B_ * P1 * L1 * 2);    // 58.7 MB
  unsigned short* adjg = (unsigned short*)carve((size_t)B_ * P1P * P1P * 2);  //  8.4 MB
  unsigned short* h2t  = (unsigned short*)carve((size_t)B_ * L2 * P1P * 2);   // 67.1 MB
  unsigned short* hB   = (unsigned short*)carve((size_t)B_ * P1 * L2 * 2);    // 58.7 MB
  float*          pooled = (float*)carve((size_t)B_ * L2 * 4);

  // 1) batch-independent stage-1 weights (diag slice + bias), f32 + bf16^T
  k_w1diag<<<(L1 * KP1) / 256, 256, 0, stream>>>(alpha1, beta1, bias1, W1df, Bt1);
  // 2) x -> bf16, K padded
  k_cvt_x<<<(B_ * NODE * KP1) / 256, 256, 0, stream>>>(x, Xbf);
  // 3) stage-1 GEMM (WMMA): M=114688, K=128, N=512; block covers 128x64
  k_gemm_s1<<<dim3(L1 / 64, (B_ * NODE) / 128), 256, 0, stream>>>(Xbf, Bt1, W1df, h1);
  // 4) pool1 scores + top-56 rank select
  k_pool1<<<B_, 128, 0, stream>>>(h1, p1w, p1b, idx1, val1);
  // 5) gather hidden (scaled) + renormalized adjacency (padded to 64)
  k_gather1<<<B_, 256, 0, stream>>>(h1, x, idx1, val1, hg, adjg);
  // 6) per-node stage-2 weights, transposed (reuses Xbf region)
  k_w2t<<<(P1 * L1 * L2) / 256, 256, 0, stream>>>(alpha2, beta2, w2t);
  // 7) zero the padded node lanes of h2t (cols 56..63 must be 0 for WMMA K-pad)
  hipMemsetAsync(h2t, 0, (size_t)B_ * L2 * P1P * 2, stream);
  // 8) stage-2 einsum (WMMA): 56 x [1024,512]@[512,512] (h2 overwrites h1 region)
  k_gemm_s2a<<<dim3(L2 / 64, B_ / 128, P1), 256, 0, stream>>>(hg, w2t, bias2, h2, h2t);
  // 9) stage-2 propagate (WMMA): per-batch [64,64]@[64,512], +h2, relu
  k_gemm_s2b<<<dim3(L2 / 128, B_), 256, 0, stream>>>(adjg, h2t, h2, hB);
  // 10) pool2 + node-mean fused
  k_pool2<<<B_, 128, 0, stream>>>(hB, p2w, p2b, pooled);
  // 11) fc head
  k_head<<<B_, L3, 0, stream>>>(pooled, fc1_w, fc1_b, fc2_w, fc2_b, out);
}
